// MHAttentionLayer_6708738916835
// MI455X (gfx1250) — matrix operand order repaired and checked by reading, not compiled, not run on hardware
//
#include <hip/hip_runtime.h>
#include <hip/hip_bf16.h>

typedef __attribute__((ext_vector_type(16))) __bf16 v16bf;
typedef __attribute__((ext_vector_type(8)))  float  v8f;
typedef __attribute__((ext_vector_type(4)))  unsigned int v4u;
typedef __attribute__((ext_vector_type(8)))  int v8i;
typedef __attribute__((ext_vector_type(4)))  int v4i;

#define EMB 1024
#define DM  64
#define NH  16
#define BS  2
#define SLEN 2048
#define MTOT (BS*SLEN)

#define LDA_PAD 36   // 16x32 bf16 A tiles, padded stride
#define LDB_PAD 72   // 32x64 bf16 tiles, padded stride (144B = 128B row + 16B TDM pad)
#define LDP_PAD 36   // per-wave 16x32 P scratch

// ---- cross-lane reductions within a 16-lane half (wave32) via ds_swizzle imm ----
template <int IMM>
static __device__ __forceinline__ float swz(float x) {
  return __int_as_float(__builtin_amdgcn_ds_swizzle(__float_as_int(x), IMM));
}
static __device__ __forceinline__ float red_max16(float x) {
  x = fmaxf(x, swz<0x041f>(x));  // xor 1
  x = fmaxf(x, swz<0x081f>(x));  // xor 2
  x = fmaxf(x, swz<0x101f>(x));  // xor 4
  x = fmaxf(x, swz<0x201f>(x));  // xor 8
  return x;
}
static __device__ __forceinline__ float red_add16(float x) {
  x += swz<0x041f>(x);
  x += swz<0x081f>(x);
  x += swz<0x101f>(x);
  x += swz<0x201f>(x);
  return x;
}

// ---- WMMA fragment loaders (contiguous k = 16*h + s order in A and B, wave32) ----
static __device__ __forceinline__ v16bf load_A_frag(const __bf16* base, int ld) {
  const int lane = threadIdx.x & 31;
  const int hl   = lane >> 4;
  const int m    = lane & 15;
  const __bf16* rp = base + m * ld;
  v16bf a;
#pragma unroll
  for (int v = 0; v < 8; ++v) {
    const int k0 = hl * 16 + 2 * v;
    a[2 * v]     = rp[k0];
    a[2 * v + 1] = rp[k0 + 1];
  }
  return a;
}

static __device__ __forceinline__ v16bf load_B_frag(const __bf16* base, int col_off, int ld) {
  const int lane = threadIdx.x & 31;
  const int hl   = lane >> 4;
  const int n    = lane & 15;
  v16bf b;
#pragma unroll
  for (int v = 0; v < 8; ++v) {
    const int k = hl * 16 + 2 * v;
    b[2 * v]     = base[k * ld + col_off + n];
    b[2 * v + 1] = base[(k + 1) * ld + col_off + n];
  }
  return b;
}

// B fragment from transposed source (rows = WMMA N dim, cols = contraction dim)
static __device__ __forceinline__ v16bf load_Bt_frag(const __bf16* base, int row_off,
                                                     int k_off, int ld) {
  const int lane = threadIdx.x & 31;
  const int hl   = lane >> 4;
  const int n    = lane & 15;
  const __bf16* rp = base + (row_off + n) * ld + k_off + hl * 16;
  v16bf b;
#pragma unroll
  for (int v = 0; v < 8; ++v) {
    b[2 * v]     = rp[2 * v];
    b[2 * v + 1] = rp[2 * v + 1];
  }
  return b;
}

// ---- Tensor Data Mover: 2D bf16 tile (rows x rowlen) -> LDS with 16B row pad ----
// D# per CDNA5 ISA: group0 {count,lds_addr,global_addr,type},
// group1 {data_size, pad cfg, tensor/tile dims, dim0 stride}. pad_interval=32 dw
// (128B row), pad_amount=4 dw (16B) -> LDS row stride 144B == LDB_PAD bf16.
static __device__ __forceinline__ void tdm_load_tile_2d(unsigned lds_off,
                                                        const __bf16* gptr,
                                                        int rows, int rowlen,
                                                        int row_stride_elems) {
  const unsigned long long ga = (unsigned long long)(uintptr_t)gptr;
  v4u g0;
  g0[0] = 1u;                                        // count=1 (valid), user mode
  g0[1] = lds_off;                                   // lds_addr (bytes)
  g0[2] = (unsigned)(ga & 0xffffffffull);            // global_addr[31:0]
  g0[3] = (unsigned)((ga >> 32) & 0x1ffffffull) | (2u << 30);  // addr[56:32] | type=2
  v8i g1;
  g1[0] = (1 << 16)            // data_size = 2 bytes
        | (1 << 20)            // pad_enable
        | (4 << 22)            // pad_interval code 4 = 32 DWORDs
        | (3 << 25);           // pad_amount  code 3 = 4 DWORDs
  g1[1] = (rowlen & 0xffff) << 16;                          // tensor_dim0[15:0]
  g1[2] = ((rowlen >> 16) & 0xffff) | ((rows & 0xffff) << 16); // dim0 hi | tensor_dim1 lo
  g1[3] = ((rows >> 16) & 0xffff) | ((rowlen & 0xffff) << 16); // dim1 hi | tile_dim0
  g1[4] = (rows & 0xffff);                                  // tile_dim1 (tile_dim2=0)
  g1[5] = row_stride_elems;                                 // tensor_dim0_stride lo32
  g1[6] = 0;
  g1[7] = 0;
  const v4i z4 = {0, 0, 0, 0};
  const v8i z8 = {0, 0, 0, 0, 0, 0, 0, 0};
  __builtin_amdgcn_tensor_load_to_lds(g0, g1, z4, z4, z8, 0);
}

// ---- GEMM + bias (+scale): out = (A@W + bias) * scale ----
// Block tile 256x64, 256 threads = 8 waves, each wave a 32x64 strip (8 WMMA/k-step).
template <bool A_BF16, bool OUT_BF16>
__global__ __launch_bounds__(256) void gemm_bias(const void* __restrict__ Ap,
                                                 const float* __restrict__ W,
                                                 const float* __restrict__ bias,
                                                 void* __restrict__ Outp,
                                                 int M, int N, int K, float scale) {
  __shared__ __bf16 sA[256 * LDA_PAD];
  __shared__ __bf16 sB[32 * LDB_PAD];

  const int mb   = blockIdx.y * 256;
  const int nb   = blockIdx.x * 64;
  const int tid  = threadIdx.x;
  const int wv   = tid >> 5;
  const int lane = tid & 31;
  const int hl   = lane >> 4;
  const int nn   = lane & 15;

  v8f acc0[4], acc1[4];
#pragma unroll
  for (int t = 0; t < 4; ++t)
#pragma unroll
    for (int r = 0; r < 8; ++r) { acc0[t][r] = 0.f; acc1[t][r] = 0.f; }

  for (int kb = 0; kb < K; kb += 32) {
    // stage A tile 256x32 (fp32 -> bf16 conversion on the fly if needed)
#pragma unroll
    for (int i = 0; i < 32; ++i) {
      const int idx = tid + i * 256;
      const int r = idx >> 5, c = idx & 31;
      __bf16 val;
      if constexpr (A_BF16)
        val = ((const __bf16*)Ap)[(size_t)(mb + r) * K + kb + c];
      else
        val = (__bf16)(((const float*)Ap)[(size_t)(mb + r) * K + kb + c]);
      sA[r * LDA_PAD + c] = val;
    }
    // stage W tile 32x64
#pragma unroll
    for (int i = 0; i < 8; ++i) {
      const int idx = tid + i * 256;
      const int r = idx >> 6, c = idx & 63;
      sB[r * LDB_PAD + c] = (__bf16)W[(size_t)(kb + r) * N + nb + c];
    }
    // prefetch next W tile into L2 (global_prefetch_b8)
    if (kb + 32 < K)
      __builtin_prefetch(&W[(size_t)(kb + 32 + (tid >> 6)) * N + nb + (tid & 63)], 0, 1);
    __syncthreads();

    const v16bf a0 = load_A_frag(&sA[(wv * 32) * LDA_PAD], LDA_PAD);
    const v16bf a1 = load_A_frag(&sA[(wv * 32 + 16) * LDA_PAD], LDA_PAD);
#pragma unroll
    for (int t = 0; t < 4; ++t) {
      const v16bf b = load_B_frag(sB, t * 16, LDB_PAD);
      acc0[t] = __builtin_amdgcn_wmma_f32_16x16x32_bf16(false, a0, false, b,
                                                        (short)0, acc0[t], false, false);
      acc1[t] = __builtin_amdgcn_wmma_f32_16x16x32_bf16(false, a1, false, b,
                                                        (short)0, acc1[t], false, false);
    }
    __syncthreads();
  }

#pragma unroll
  for (int t = 0; t < 4; ++t) {
#pragma unroll
    for (int r = 0; r < 8; ++r) {
      const int col = nb + t * 16 + nn;
      const int row0 = mb + wv * 32 + r + 8 * hl;
      const int row1 = row0 + 16;
      const float v0 = (acc0[t][r] + bias[col]) * scale;
      const float v1 = (acc1[t][r] + bias[col]) * scale;
      if constexpr (OUT_BF16) {
        ((__bf16*)Outp)[(size_t)row0 * N + col] = (__bf16)v0;
        ((__bf16*)Outp)[(size_t)row1 * N + col] = (__bf16)v1;
      } else {
        ((float*)Outp)[(size_t)row0 * N + col] = v0;
        ((float*)Outp)[(size_t)row1 * N + col] = v1;
      }
    }
  }
}

// ---- Flash attention: per (batch, head, 128 q rows); wave owns 16 q rows ----
__global__ __launch_bounds__(256) void attn(const __bf16* __restrict__ qb,
                                            const __bf16* __restrict__ kb,
                                            const __bf16* __restrict__ vb,
                                            __bf16* __restrict__ ob) {
  __shared__ __bf16 sK[32 * LDB_PAD];
  __shared__ __bf16 sV[32 * LDB_PAD];
  __shared__ __bf16 sP[8 * 16 * LDP_PAD];  // per-wave P transpose scratch

  const int tid  = threadIdx.x;
  const int wv   = tid >> 5;
  const int lane = tid & 31;
  const int hl   = lane >> 4;
  const int nn   = lane & 15;

  const int nbatch = blockIdx.z;
  const int head   = blockIdx.y;
  const int q0     = blockIdx.x * 128 + wv * 16;

  // preload Q fragments (1/sqrt(d) already folded in by the projection pass)
  const __bf16* qptr = qb + ((size_t)nbatch * SLEN + q0) * EMB + head * DM;
  const v16bf aq0 = load_A_frag(qptr, EMB);
  const v16bf aq1 = load_A_frag(qptr + 32, EMB);

  v8f O[4];
  float mrow[8], lrow[8];
#pragma unroll
  for (int t = 0; t < 4; ++t)
#pragma unroll
    for (int r = 0; r < 8; ++r) O[t][r] = 0.f;
#pragma unroll
  for (int r = 0; r < 8; ++r) { mrow[r] = -3.0e38f; lrow[r] = 0.f; }

  __bf16* myP = &sP[wv * 16 * LDP_PAD];
  const unsigned sK_off = (unsigned)(uintptr_t)sK;  // LDS byte offset (low 32 bits)
  const unsigned sV_off = (unsigned)(uintptr_t)sV;

  for (int j0 = 0; j0 < SLEN; j0 += 32) {
    // TDM: one wave DMAs the 32x64 bf16 K and V tiles into padded LDS
    if (wv == 0) {
      const size_t g = ((size_t)nbatch * SLEN + j0) * EMB + head * DM;
      tdm_load_tile_2d(sK_off, kb + g, 32, 64, EMB);
      tdm_load_tile_2d(sV_off, vb + g, 32, 64, EMB);
      __builtin_amdgcn_s_wait_tensorcnt(0);
    }
    __syncthreads();

    // S = q @ K^T : two 16x16 S subtiles, contraction d=64 in two k-steps
    v8f s0 = {0, 0, 0, 0, 0, 0, 0, 0};
    v8f s1 = {0, 0, 0, 0, 0, 0, 0, 0};
    s0 = __builtin_amdgcn_wmma_f32_16x16x32_bf16(false, aq0, false,
             load_Bt_frag(sK, 0, 0, LDB_PAD), (short)0, s0, false, false);
    s0 = __builtin_amdgcn_wmma_f32_16x16x32_bf16(false, aq1, false,
             load_Bt_frag(sK, 0, 32, LDB_PAD), (short)0, s0, false, false);
    s1 = __builtin_amdgcn_wmma_f32_16x16x32_bf16(false, aq0, false,
             load_Bt_frag(sK, 16, 0, LDB_PAD), (short)0, s1, false, false);
    s1 = __builtin_amdgcn_wmma_f32_16x16x32_bf16(false, aq1, false,
             load_Bt_frag(sK, 16, 32, LDB_PAD), (short)0, s1, false, false);

    // online softmax; row lives across a 16-lane half -> ds_swizzle xor reduce
    float alpha[8];
#pragma unroll
    for (int r = 0; r < 8; ++r) {
      const float mx = red_max16(fmaxf(s0[r], s1[r]));
      const float mn = fmaxf(mrow[r], mx);
      alpha[r] = __expf(mrow[r] - mn);
      mrow[r] = mn;
      s0[r] = __expf(s0[r] - mn);
      s1[r] = __expf(s1[r] - mn);
      lrow[r] = lrow[r] * alpha[r] + red_add16(s0[r] + s1[r]);
    }
#pragma unroll
    for (int t = 0; t < 4; ++t)
#pragma unroll
      for (int r = 0; r < 8; ++r) O[t][r] *= alpha[r];

    // transpose P (D layout -> A layout) through per-wave LDS scratch
#pragma unroll
    for (int r = 0; r < 8; ++r) {
      myP[(r + 8 * hl) * LDP_PAD + nn]      = (__bf16)s0[r];
      myP[(r + 8 * hl) * LDP_PAD + 16 + nn] = (__bf16)s1[r];
    }
    asm volatile("s_wait_dscnt 0" ::: "memory");
    const v16bf ap = load_A_frag(myP, LDP_PAD);

    // O += P @ V (contraction over 32 kv rows, 4 d-column groups)
#pragma unroll
    for (int t = 0; t < 4; ++t) {
      const v16bf bfv = load_B_frag(sV, t * 16, LDB_PAD);
      O[t] = __builtin_amdgcn_wmma_f32_16x16x32_bf16(false, ap, false, bfv,
                                                     (short)0, O[t], false, false);
    }
    __syncthreads();  // all waves done with sK/sV before next TDM overwrite
  }

  // normalize and store bf16 attention output at (n, l, h*DM + d)
  float inv[8];
#pragma unroll
  for (int r = 0; r < 8; ++r) inv[r] = 1.0f / lrow[r];
#pragma unroll
  for (int t = 0; t < 4; ++t)
#pragma unroll
    for (int r = 0; r < 8; ++r) {
      const int row = q0 + r + 8 * hl;
      ob[((size_t)nbatch * SLEN + row) * EMB + head * DM + t * 16 + nn] =
          (__bf16)(O[t][r] * inv[r]);
    }
}

extern "C" void kernel_launch(void* const* d_in, const int* in_sizes, int n_in,
                              void* d_out, int out_size, void* d_ws, size_t ws_size,
                              hipStream_t stream) {
  (void)in_sizes; (void)n_in; (void)out_size; (void)ws_size;
  const float* Q  = (const float*)d_in[0];
  const float* K  = (const float*)d_in[1];
  const float* V  = (const float*)d_in[2];
  const float* Wq = (const float*)d_in[3];
  const float* bq = (const float*)d_in[4];
  const float* Wk = (const float*)d_in[5];
  const float* bk = (const float*)d_in[6];
  const float* Wv = (const float*)d_in[7];
  const float* bv = (const float*)d_in[8];
  const float* Wo = (const float*)d_in[9];
  const float* bo = (const float*)d_in[10];

  const size_t nelem = (size_t)MTOT * EMB;  // 4M elements
  __bf16* qb = (__bf16*)d_ws;
  __bf16* kb = qb + nelem;
  __bf16* vb = kb + nelem;
  __bf16* ob = vb + nelem;

  const float scale = 0.125f;  // 1/sqrt(DM)

  const dim3 blk(256);
  const dim3 gproj(EMB / 64, MTOT / 256);  // 16 x 16
  gemm_bias<false, true><<<gproj, blk, 0, stream>>>(Q, Wq, bq, qb, MTOT, EMB, EMB, scale);
  gemm_bias<false, true><<<gproj, blk, 0, stream>>>(K, Wk, bk, kb, MTOT, EMB, EMB, 1.0f);
  gemm_bias<false, true><<<gproj, blk, 0, stream>>>(V, Wv, bv, vb, MTOT, EMB, EMB, 1.0f);

  const dim3 gattn(SLEN / 128, NH, BS);    // 16 x 16 x 2
  attn<<<gattn, blk, 0, stream>>>(qb, kb, vb, ob);

  gemm_bias<true, false><<<gproj, blk, 0, stream>>>(ob, Wo, bo, d_out, MTOT, EMB, EMB, 1.0f);
}
